// InpaintGenerator_42640435315394
// MI455X (gfx1250) — matrix operand order repaired and checked by reading, not compiled, not run on hardware
//
#include <hip/hip_runtime.h>
#include <hip/hip_bf16.h>
#include <math.h>
#include <stdint.h>

#define USE_ASYNC_LDS 1   // CDNA5 global_load_async_to_lds_b128 + s_wait_asynccnt

// ---------------------------------------------------------------------------
// Types for CDNA5 WMMA (wave32): D(16x16 f32) = A(16x32 bf16) * B(32x16 bf16) + C
// ---------------------------------------------------------------------------
typedef __bf16 bf16_t;
typedef __attribute__((ext_vector_type(16))) __bf16        v16bf;
typedef __attribute__((ext_vector_type(8)))  float         v8f;
typedef __attribute__((ext_vector_type(4)))  unsigned int  v4u;

union FragBF { v4u q[2]; v16bf v; };
union F8     { v4u q[2]; float f[8]; };

// Fragment loads as uniform-base + 32-bit lane index (SADDR+voffset friendly).
// A-matrix 16x32 bf16: lane's 16 contiguous K values are vidx and vidx+1.
__device__ __forceinline__ v16bf load_a_frag_i(const v4u* base, int vidx) {
  FragBF f;
  f.q[0] = base[vidx];
  f.q[1] = base[vidx + 1];
  return f.v;
}
// B-matrix 32x16 bf16: lane's 16 contiguous K values are vidx and vidx+1.
__device__ __forceinline__ v16bf load_b_frag_i(const v4u* base, int vidx) {
  FragBF f;
  f.q[0] = base[vidx];
  f.q[1] = base[vidx + 1];
  return f.v;
}
// LDS variant (pointer form; ds addressing with immediate offsets is optimal)
__device__ __forceinline__ v16bf load_b_frag(const bf16_t* colk0, int laneHi) {
  FragBF f;
  const v4u* p = (const v4u*)(colk0 + laneHi * 16);
  f.q[0] = p[0];
  f.q[1] = p[1];
  return f.v;
}
__device__ __forceinline__ v8f wmma_bf16(v16bf a, v16bf b, v8f c) {
  return __builtin_amdgcn_wmma_f32_16x16x32_bf16(
      false, a, false, b, (short)0, c, false, false);
}

// ---------------------------------------------------------------------------
#define NB   8      // images (bsz*t)
#define NC   128    // channels
#define HW   4096   // 64*64
#define WID  64

// ---------------------------------------------------------------------------
// init: xf (NHWC f32 master in ws) + xbf (NHWC bf16 working copy)
// ---------------------------------------------------------------------------
__global__ __launch_bounds__(256) void k_init_x(const float* __restrict__ x,
    float* __restrict__ xf, bf16_t* __restrict__ xb, int n) {
  int i = blockIdx.x * 256 + threadIdx.x;          // NCHW flat index
  if (i >= n) return;
  float v = x[i];
  int img = i >> 19;            // NC*HW = 2^19
  int rem = i & ((1 << 19) - 1);
  int c = rem >> 12;            // HW = 2^12
  int p = rem & (HW - 1);
  size_t nh = ((size_t)img * HW + p) * NC + c;
  xf[nh] = v;
  xb[nh] = (bf16_t)v;
}

// final: NHWC f32 master -> NCHW f32 d_out
__global__ __launch_bounds__(256) void k_final(const float* __restrict__ xf,
    float* __restrict__ out, int n) {
  int i = blockIdx.x * 256 + threadIdx.x;          // NCHW flat index
  if (i >= n) return;
  int img = i >> 19;
  int rem = i & ((1 << 19) - 1);
  int c = rem >> 12;
  int p = rem & (HW - 1);
  out[i] = xf[((size_t)img * HW + p) * NC + c];
}

// 1x1-conv weights (3 tensors, each (2,128,128)) to bf16, layout kept row-major
__global__ __launch_bounds__(256) void k_cvt_lin(const float* __restrict__ wq,
    const float* __restrict__ wk, const float* __restrict__ wv,
    bf16_t* __restrict__ out) {
  int i = blockIdx.x * 256 + threadIdx.x;          // < 98304
  int which = i >> 15; int rem = i & 32767;
  const float* s = (which == 0) ? wq : (which == 1) ? wk : wv;
  out[i] = (bf16_t)s[rem];
}

// 3x3-conv weights (3 tensors, each (2,128,128,3,3)) -> bf16 tap-major
__global__ __launch_bounds__(256) void k_cvt_conv(const float* __restrict__ wo,
    const float* __restrict__ wf1, const float* __restrict__ wf2,
    bf16_t* __restrict__ out) {
  int i = blockIdx.x * 256 + threadIdx.x;          // < 884736
  int which = i / 294912; int rem = i % 294912;
  int layer = rem / 147456; int r2 = rem % 147456;
  int tap = r2 >> 14; int oi = r2 & 16383;
  const float* s = (which == 0) ? wo : (which == 1) ? wf1 : wf2;
  out[i] = (bf16_t)s[layer * 147456 + oi * 9 + tap];
}

// ---------------------------------------------------------------------------
// conv-as-GEMM (taps=1 -> 1x1).  Activations NHWC bf16; f32 master NHWC.
// Block 256 thr (8 waves): M=128 out-channels x N=128 pixels, K=128 in steps
// of 32, accumulated over taps.  LDS tile [n][k] double-buffered, staged with
// async global->LDS b128 copies (one per-pixel predicate, zero-fill borders).
// mode: 0 = +bias -> bf16                  (QKV)
//       1 = leaky(+bias) -> bf16           (Wf1)
//       2 = resid + leaky(+bias) -> f32 NHWC + bf16 NHWC (Wo/Wf2)
// ---------------------------------------------------------------------------
__global__ __launch_bounds__(256) void k_gemm_conv(
    const bf16_t* __restrict__ inbf, const bf16_t* __restrict__ wt,
    const float* __restrict__ bias, int taps, int dil, int mode,
    float* __restrict__ outf, bf16_t* __restrict__ outb,
    const float* __restrict__ resid)
{
  __shared__ alignas(16) bf16_t Bt[2][128 * 40];   // [n][k], padded stride 40

  const int tid = threadIdx.x;
  const int lane = tid & 31, wave = tid >> 5;
  const int lane16 = lane & 15, laneHi = lane >> 4;
  const int img = blockIdx.y;
  const int p0 = blockIdx.x * 128;
  const bf16_t* xb = inbf + (size_t)img * HW * NC;   // NHWC image base
  const int m0 = wave * 16;

  // staging identity: pixel sn, channel half skh
  const int sn = tid & 127;
  const int skh = tid >> 7;
  const int sp = p0 + sn;
  const int sh = sp >> 6, sw = sp & 63;

  const int nsteps = taps * 4;

  auto stage = [&](int s, int buf) {
    int tap = s >> 2;
    int k0 = (s & 3) * 32;
    int t3 = (tap * 11) >> 5;                  // tap / 3 for 0..8
    int dh = (taps == 1) ? 0 : dil * (t3 - 1);
    int dw = (taps == 1) ? 0 : dil * (tap - 3 * t3 - 1);
    int h = sh + dh, w = sw + dw;
    bf16_t* dst = &Bt[buf][sn * 40 + skh * 16];
    if (h >= 0 && h < WID && w >= 0 && w < WID) {
      const bf16_t* src = xb + (size_t)(h * WID + w) * NC + k0 + skh * 16;
#if USE_ASYNC_LDS
      asm volatile(
          "global_load_async_to_lds_b128 %0, %2, off\n\t"
          "global_load_async_to_lds_b128 %1, %3, off"
          :: "v"((unsigned int)(uintptr_t)dst),
             "v"((unsigned int)((uintptr_t)dst + 16)),
             "v"(src), "v"(src + 8)
          : "memory");
#else
      v4u a = *(const v4u*)src;
      v4u b = *(const v4u*)(src + 8);
      *(v4u*)dst = a;
      *(v4u*)(dst + 8) = b;
#endif
    } else {
      v4u z = {0u, 0u, 0u, 0u};
      *(v4u*)dst = z;
      *(v4u*)(dst + 8) = z;
    }
  };

  const v8f vz = {0.f, 0.f, 0.f, 0.f, 0.f, 0.f, 0.f, 0.f};
  v8f acc[8];
#pragma unroll
  for (int i = 0; i < 8; ++i) acc[i] = vz;

  const int avidx = (m0 + lane16) * 16 + laneHi * 2;   // 16 v4u per 128-bf16 row

  stage(0, 0);
  for (int s = 0; s < nsteps; ++s) {
    const int buf = s & 1;
#if USE_ASYNC_LDS
    asm volatile("s_wait_asynccnt 0x0" ::: "memory");
#endif
    __syncthreads();
    if (s + 1 < nsteps) stage(s + 1, buf ^ 1);   // prefetch next tile

    const int tap = s >> 2, k0 = (s & 3) * 32;
    const v4u* wb = (const v4u*)(wt + (size_t)tap * 16384 + k0);  // uniform
    __builtin_prefetch(wb + avidx + 16, 0, 3);   // next row of weights
    v16bf afrag = load_a_frag_i(wb, avidx);
#pragma unroll
    for (int nt = 0; nt < 8; ++nt) {
      v16bf bfrag = load_b_frag(&Bt[buf][(nt * 16 + lane16) * 40], laneHi);
      acc[nt] = wmma_bf16(afrag, bfrag, acc[nt]);
    }
  }

  // epilogue: lane owns 8 consecutive channels (m0+laneHi*8 .. +7) at pixel n
  F8 bv;
  bv.q[0] = *(const v4u*)(bias + m0 + laneHi * 8);
  bv.q[1] = *(const v4u*)(bias + m0 + laneHi * 8 + 4);
#pragma unroll
  for (int nt = 0; nt < 8; ++nt) {
    int n = p0 + nt * 16 + lane16;
    size_t nb = ((size_t)img * HW + n) * NC + m0 + laneHi * 8;  // NHWC
    union { bf16_t h[8]; v4u q; } pk;
    if (mode == 2) {
      F8 rv, ov;
      rv.q[0] = *(const v4u*)(resid + nb);
      rv.q[1] = *(const v4u*)(resid + nb + 4);
#pragma unroll
      for (int r = 0; r < 8; ++r) {
        float v = acc[nt][r] + bv.f[r];
        v = (v > 0.f) ? v : 0.2f * v;
        float o = rv.f[r] + v;
        ov.f[r] = o;
        pk.h[r] = (bf16_t)o;
      }
      *(v4u*)(outf + nb) = ov.q[0];
      *(v4u*)(outf + nb + 4) = ov.q[1];
    } else {
#pragma unroll
      for (int r = 0; r < 8; ++r) {
        float v = acc[nt][r] + bv.f[r];
        if (mode != 0) v = (v > 0.f) ? v : 0.2f * v;
        pk.h[r] = (bf16_t)v;
      }
    }
    *(v4u*)(outb + nb) = pk.q;
  }
}

// ---------------------------------------------------------------------------
// Repack q/k/v (NHWC bf16) -> Qt,Kt (L x d) and VtT (d x L).
// All group dims are powers of two: psh=log2(ps), ohsh=log2(oh), dsh=log2(d).
// ---------------------------------------------------------------------------
__global__ __launch_bounds__(256) void k_repack(
    const bf16_t* __restrict__ qb, const bf16_t* __restrict__ kb,
    const bf16_t* __restrict__ vb,
    bf16_t* __restrict__ Qt, bf16_t* __restrict__ Kt, bf16_t* __restrict__ VtT,
    int L, int g, int bb, int psh, int ohsh, int dsh)
{
  int i = blockIdx.x * 256 + threadIdx.x;       // i < L*d = 524288
  int l = i >> dsh, dd = i & ((1 << dsh) - 1);
  int c  = dd >> (2 * psh);
  int r2 = dd & ((1 << (2 * psh)) - 1);
  int ih = r2 >> psh, iw = r2 & ((1 << psh) - 1);
  int tt = l >> (2 * ohsh);
  int rl = l & ((1 << (2 * ohsh)) - 1);
  int ohh = rl >> ohsh, oww = rl & ((1 << ohsh) - 1);
  int pix = ((ohh << psh) + ih) * WID + ((oww << psh) + iw);
  size_t src = ((size_t)(bb * 4 + tt) * HW + pix) * NC + g * 32 + c;  // NHWC
  Qt[i] = qb[src];
  Kt[i] = kb[src];
  VtT[((size_t)dd << (dsh == 7 ? 12 : (19 - dsh))) + l] = vb[src];
  // note: L = 2^(19-dsh) since L*d = 2^19
}

// ---------------------------------------------------------------------------
// scores = (Qt @ Kt^T) * scale   (M=N=L, K=d).  Block tile 64M x 128N, 8 waves.
// ---------------------------------------------------------------------------
__global__ __launch_bounds__(256) void k_scores(
    const bf16_t* __restrict__ Qt, const bf16_t* __restrict__ Kt,
    float* __restrict__ S, int L, int d, float scale)
{
  int tid = threadIdx.x, lane = tid & 31, wave = tid >> 5;
  int lane16 = lane & 15, laneHi = lane >> 4;
  int m0 = blockIdx.x * 64 + (wave & 3) * 16;
  int n0 = blockIdx.y * 128 + (wave >> 2) * 64;
  if (n0 >= L) return;                         // L==64: waves 4-7 idle
  const v8f vz = {0.f, 0.f, 0.f, 0.f, 0.f, 0.f, 0.f, 0.f};
  v8f acc[4] = {vz, vz, vz, vz};
  const int rowq = d >> 3;                     // v4u per row
  const int avidx = (m0 + lane16) * rowq + laneHi * 2;
  int bvidx[4];
#pragma unroll
  for (int nt = 0; nt < 4; ++nt)
    bvidx[nt] = (n0 + nt * 16 + lane16) * rowq + laneHi * 2;
  for (int k0 = 0; k0 < d; k0 += 32) {
    const v4u* qb4 = (const v4u*)(Qt + k0);
    const v4u* kb4 = (const v4u*)(Kt + k0);
    v16bf a = load_a_frag_i(qb4, avidx);
#pragma unroll
    for (int nt = 0; nt < 4; ++nt)
      acc[nt] = wmma_bf16(a, load_b_frag_i(kb4, bvidx[nt]), acc[nt]);
  }
#pragma unroll
  for (int nt = 0; nt < 4; ++nt)
#pragma unroll
    for (int r = 0; r < 8; ++r)
      S[(size_t)(m0 + r + laneHi * 8) * L + n0 + nt * 16 + lane16] =
          acc[nt][r] * scale;
}

// ---------------------------------------------------------------------------
// Row softmax over L; writes bf16 probabilities P.
// ---------------------------------------------------------------------------
__global__ __launch_bounds__(256) void k_softmax(
    float* __restrict__ S, bf16_t* __restrict__ P, int L)
{
  __shared__ float red[8];
  int row = blockIdx.x, tid = threadIdx.x;
  int lane = tid & 31, wave = tid >> 5;
  float* s = S + (size_t)row * L;

  float mx = -3.4e38f;
  for (int i = tid; i < L; i += 256) mx = fmaxf(mx, s[i]);
  for (int o = 16; o; o >>= 1) mx = fmaxf(mx, __shfl_xor(mx, o, 32));
  if (lane == 0) red[wave] = mx;
  __syncthreads();
  mx = red[0];
  for (int w = 1; w < 8; ++w) mx = fmaxf(mx, red[w]);
  __syncthreads();

  float sum = 0.f;
  for (int i = tid; i < L; i += 256) {
    float e = __expf(s[i] - mx);
    s[i] = e;
    sum += e;
  }
  for (int o = 16; o; o >>= 1) sum += __shfl_xor(sum, o, 32);
  if (lane == 0) red[wave] = sum;
  __syncthreads();
  sum = 0.f;
  for (int w = 0; w < 8; ++w) sum += red[w];
  float inv = 1.0f / sum;

  for (int i = tid; i < L; i += 256)
    P[(size_t)row * L + i] = (bf16_t)(s[i] * inv);
}

// ---------------------------------------------------------------------------
// Yt = P @ Vt (M=L, N=d, K=L); epilogue scatters bf16 into att (NHWC).
// ---------------------------------------------------------------------------
__global__ __launch_bounds__(256) void k_pv(
    const bf16_t* __restrict__ P, const bf16_t* __restrict__ VtT,
    bf16_t* __restrict__ attb, int L, int g, int bb, int psh, int ohsh)
{
  int tid = threadIdx.x, lane = tid & 31, wave = tid >> 5;
  int lane16 = lane & 15, laneHi = lane >> 4;
  int m0 = blockIdx.x * 64 + (wave & 3) * 16;
  int n0 = blockIdx.y * 128 + (wave >> 2) * 64;
  const v8f vz = {0.f, 0.f, 0.f, 0.f, 0.f, 0.f, 0.f, 0.f};
  v8f acc[4] = {vz, vz, vz, vz};
  const int rowq = L >> 3;                     // v4u per row
  const int avidx = (m0 + lane16) * rowq + laneHi * 2;
  int bvidx[4];
#pragma unroll
  for (int nt = 0; nt < 4; ++nt)
    bvidx[nt] = (n0 + nt * 16 + lane16) * rowq + laneHi * 2;
  for (int k0 = 0; k0 < L; k0 += 32) {
    const v4u* pb4 = (const v4u*)(P + k0);
    const v4u* vb4 = (const v4u*)(VtT + k0);
    v16bf a = load_a_frag_i(pb4, avidx);
#pragma unroll
    for (int nt = 0; nt < 4; ++nt)
      acc[nt] = wmma_bf16(a, load_b_frag_i(vb4, bvidx[nt]), acc[nt]);
  }
#pragma unroll
  for (int nt = 0; nt < 4; ++nt) {
#pragma unroll
    for (int r = 0; r < 8; ++r) {
      int l = m0 + r + laneHi * 8;
      int dd = n0 + nt * 16 + lane16;
      int c  = dd >> (2 * psh);
      int r2 = dd & ((1 << (2 * psh)) - 1);
      int ih = r2 >> psh, iw = r2 & ((1 << psh) - 1);
      int tt = l >> (2 * ohsh);
      int rl = l & ((1 << (2 * ohsh)) - 1);
      int ohh = rl >> ohsh, oww = rl & ((1 << ohsh) - 1);
      int pix = ((ohh << psh) + ih) * WID + ((oww << psh) + iw);
      attb[((size_t)(bb * 4 + tt) * HW + pix) * NC + g * 32 + c] =
          (bf16_t)acc[nt][r];
    }
  }
}

// ---------------------------------------------------------------------------
// Host-side orchestration
// ---------------------------------------------------------------------------
extern "C" void kernel_launch(void* const* d_in, const int* in_sizes, int n_in,
                              void* d_out, int out_size, void* d_ws, size_t ws_size,
                              hipStream_t stream) {
  (void)in_sizes; (void)n_in; (void)out_size; (void)ws_size;
  const float* x   = (const float*)d_in[0];
  const float* Wq  = (const float*)d_in[1];  const float* bq  = (const float*)d_in[2];
  const float* Wk  = (const float*)d_in[3];  const float* bk  = (const float*)d_in[4];
  const float* Wv  = (const float*)d_in[5];  const float* bv  = (const float*)d_in[6];
  const float* Wo  = (const float*)d_in[7];  const float* bo  = (const float*)d_in[8];
  const float* Wf1 = (const float*)d_in[9];  const float* bf1 = (const float*)d_in[10];
  const float* Wf2 = (const float*)d_in[11]; const float* bf2 = (const float*)d_in[12];

  char* ws = (char*)d_ws;
  const size_t SZ_ACT = (size_t)NB * NC * HW * sizeof(bf16_t);  // 8 MB
  bf16_t* xbf   = (bf16_t*)(ws);
  bf16_t* qbf   = (bf16_t*)(ws + 1 * SZ_ACT);
  bf16_t* kbf   = (bf16_t*)(ws + 2 * SZ_ACT);
  bf16_t* vbf   = (bf16_t*)(ws + 3 * SZ_ACT);
  bf16_t* attbf = (bf16_t*)(ws + 4 * SZ_ACT);
  bf16_t* fbf   = (bf16_t*)(ws + 5 * SZ_ACT);
  size_t off = 6 * SZ_ACT;
  bf16_t* wlin = (bf16_t*)(ws + off); off += (size_t)98304 * 2;
  bf16_t* wcv  = (bf16_t*)(ws + off); off += (size_t)884736 * 2;
  bf16_t* Qt   = (bf16_t*)(ws + off); off += (size_t)524288 * 2;
  bf16_t* Kt   = (bf16_t*)(ws + off); off += (size_t)524288 * 2;
  bf16_t* VtT  = (bf16_t*)(ws + off); off += (size_t)524288 * 2;
  float*  xf   = (float*)(ws + off);  off += (size_t)NB * NC * HW * 4;  // NHWC f32
  bf16_t* Pbf  = (bf16_t*)(ws + off); off += (size_t)4096 * 4096 * 2;
  float*  Sc   = (float*)(ws + off);

  k_init_x<<<16384, 256, 0, stream>>>(x, xf, xbf, NB * NC * HW);
  k_cvt_lin<<<384, 256, 0, stream>>>(Wq, Wk, Wv, wlin);
  k_cvt_conv<<<3456, 256, 0, stream>>>(Wo, Wf1, Wf2, wcv);

  static const int PSH[4]  = {1, 2, 3, 4};   // log2 patch
  static const int OHSH[4] = {5, 4, 3, 2};   // log2 (64/patch)
  const dim3 gg(32, 8);   // 32 pixel-tiles of 128, 8 images

  for (int layer = 0; layer < 2; ++layer) {
    const bf16_t* wq_l = wlin + (size_t)(0 * 2 + layer) * 16384;
    const bf16_t* wk_l = wlin + (size_t)(1 * 2 + layer) * 16384;
    const bf16_t* wv_l = wlin + (size_t)(2 * 2 + layer) * 16384;
    k_gemm_conv<<<gg, 256, 0, stream>>>(xbf, wq_l, bq + layer * NC, 1, 1, 0,
                                        nullptr, qbf, nullptr);
    k_gemm_conv<<<gg, 256, 0, stream>>>(xbf, wk_l, bk + layer * NC, 1, 1, 0,
                                        nullptr, kbf, nullptr);
    k_gemm_conv<<<gg, 256, 0, stream>>>(xbf, wv_l, bv + layer * NC, 1, 1, 0,
                                        nullptr, vbf, nullptr);

    for (int g = 0; g < 4; ++g) {
      int psh = PSH[g], ohsh = OHSH[g];
      int oh = 1 << ohsh;
      int L = 4 * oh * oh, d = 32 << (2 * psh);
      int dsh = 5 + 2 * psh;
      float scale = 1.0f / sqrtf((float)d);
      for (int bb = 0; bb < 2; ++bb) {
        k_repack<<<2048, 256, 0, stream>>>(qbf, kbf, vbf, Qt, Kt, VtT,
                                           L, g, bb, psh, ohsh, dsh);
        dim3 gs(L / 64, (L + 127) / 128);
        k_scores<<<gs, 256, 0, stream>>>(Qt, Kt, Sc, L, d, scale);
        k_softmax<<<L, 256, 0, stream>>>(Sc, Pbf, L);
        dim3 gp(L / 64, d / 128);
        k_pv<<<gp, 256, 0, stream>>>(Pbf, VtT, attbf, L, g, bb, psh, ohsh);
      }
    }

    const bf16_t* wo_l  = wcv + (size_t)(0 * 2 + layer) * 147456;
    const bf16_t* wf1_l = wcv + (size_t)(1 * 2 + layer) * 147456;
    const bf16_t* wf2_l = wcv + (size_t)(2 * 2 + layer) * 147456;
    // x = x + leaky(conv3x3(att))
    k_gemm_conv<<<gg, 256, 0, stream>>>(attbf, wo_l, bo + layer * NC, 9, 1, 2,
                                        xf, xbf, xf);
    // f = leaky(conv3x3_dil2(x))
    k_gemm_conv<<<gg, 256, 0, stream>>>(xbf, wf1_l, bf1 + layer * NC, 9, 2, 1,
                                        nullptr, fbf, nullptr);
    // x = x + leaky(conv3x3(f))
    k_gemm_conv<<<gg, 256, 0, stream>>>(fbf, wf2_l, bf2 + layer * NC, 9, 1, 2,
                                        xf, xbf, xf);
  }

  k_final<<<16384, 256, 0, stream>>>(xf, (float*)d_out, NB * NC * HW);
}
